// DebertaV2MoELayer_84387517432150
// MI455X (gfx1250) — matrix operand order repaired and checked by reading, not compiled, not run on hardware
//
#include <hip/hip_runtime.h>
#include <hip/hip_bf16.h>
#include <math.h>

typedef __bf16 bf16_t;
typedef __attribute__((ext_vector_type(16))) __bf16 v16bf;
typedef __attribute__((ext_vector_type(8)))  __bf16 v8bf;
typedef __attribute__((ext_vector_type(8)))  float  v8f;
typedef __attribute__((ext_vector_type(4)))  float  v4f;

#define T_TOK 4096
#define H_DIM 1024
#define I_DIM 4096
#define E_EXP 8
#define K_TOP 2
#define NSLOT (T_TOK * K_TOP)
#define LN_EPS 1e-7f

// ---------------------------------------------------------------- utilities
__global__ __launch_bounds__(256) void cvt_bf16_kernel(const float* __restrict__ x,
                                                       bf16_t* __restrict__ xb) {
  size_t i = ((size_t)blockIdx.x * 256 + threadIdx.x) * 4;
  #pragma unroll
  for (int j = 0; j < 4; j++) xb[i + j] = (bf16_t)x[i + j];
}

// ---------------------------------------------------------------- router
// one wave (32 lanes) per token: logits -> softmax -> top2 -> renorm
__global__ __launch_bounds__(256) void router_kernel(
    const float* __restrict__ x, const float* __restrict__ rw,
    float* __restrict__ logits_out, int* __restrict__ sel,
    float* __restrict__ wnorm, int* __restrict__ counts) {
  const int wid = threadIdx.x >> 5;
  const int lane = threadIdx.x & 31;
  const int t = blockIdx.x * 8 + wid;
  float acc[E_EXP];
  #pragma unroll
  for (int e = 0; e < E_EXP; e++) acc[e] = 0.f;
  const float* xr = x + (size_t)t * H_DIM;
  for (int h = lane; h < H_DIM; h += 32) {
    const float xv = xr[h];
    const float* rwr = rw + (size_t)h * E_EXP;
    #pragma unroll
    for (int e = 0; e < E_EXP; e++) acc[e] += xv * rwr[e];
  }
  #pragma unroll
  for (int e = 0; e < E_EXP; e++)
    for (int m = 16; m >= 1; m >>= 1) acc[e] += __shfl_xor(acc[e], m, 32);
  float mx = acc[0];
  #pragma unroll
  for (int e = 1; e < E_EXP; e++) mx = fmaxf(mx, acc[e]);
  float p[E_EXP], sum = 0.f;
  #pragma unroll
  for (int e = 0; e < E_EXP; e++) { p[e] = __expf(acc[e] - mx); sum += p[e]; }
  const float inv = 1.f / sum;
  #pragma unroll
  for (int e = 0; e < E_EXP; e++) p[e] *= inv;
  int i0 = 0;
  #pragma unroll
  for (int e = 1; e < E_EXP; e++) if (p[e] > p[i0]) i0 = e;
  int i1 = (i0 == 0) ? 1 : 0;
  #pragma unroll
  for (int e = 0; e < E_EXP; e++) if (e != i0 && p[e] > p[i1]) i1 = e;
  const float s2 = p[i0] + p[i1];
  if (lane == 0) {
    #pragma unroll
    for (int e = 0; e < E_EXP; e++) logits_out[(size_t)t * E_EXP + e] = acc[e];
    sel[t * 2 + 0] = i0;            sel[t * 2 + 1] = i1;
    wnorm[t * 2 + 0] = p[i0] / s2;  wnorm[t * 2 + 1] = p[i1] / s2;
    atomicAdd(&counts[i0], 1);      atomicAdd(&counts[i1], 1);
  }
}

__global__ void scan_kernel(const int* __restrict__ counts, int* __restrict__ offsets) {
  if (threadIdx.x == 0) {
    int s = 0;
    for (int e = 0; e < E_EXP; e++) { offsets[e] = s; s += counts[e]; }
  }
}

__global__ __launch_bounds__(256) void assign_kernel(
    const int* __restrict__ sel, const int* __restrict__ offsets,
    int* __restrict__ cursor, int* __restrict__ list, int* __restrict__ slot_of) {
  const int t = blockIdx.x * 256 + threadIdx.x;
  #pragma unroll
  for (int j = 0; j < K_TOP; j++) {
    const int e = sel[t * 2 + j];
    const int pos = atomicAdd(&cursor[e], 1);
    const int slot = offsets[e] + pos;
    list[slot] = t;
    slot_of[t * 2 + j] = slot;
  }
}

// ---------------------------------------------------------------- WMMA fragment helpers
// A (16x32 bf16): lane L holds row M=L&15; K = kb..kb+7 and kb+16..kb+23, kb=(L>>4)*8
// B (32x16 bf16): lane L holds col N=L&15; K = kb..kb+7 and kb+16..kb+23, kb=(L>>4)*8
__device__ inline v16bf load_a_frag(const bf16_t* base, int stride, int row, int lane) {
  const int kb = (lane >> 4) * 8;
  const v8bf lo = *(const v8bf*)(base + row * stride + kb);
  const v8bf hi = *(const v8bf*)(base + row * stride + kb + 16);
  return __builtin_shufflevector(lo, hi, 0,1,2,3,4,5,6,7,8,9,10,11,12,13,14,15);
}
__device__ inline v16bf load_b_frag(const bf16_t* baseT, int stride, int col, int lane) {
  const int kb = (lane >> 4) * 8;
  const v8bf lo = *(const v8bf*)(baseT + col * stride + kb);
  const v8bf hi = *(const v8bf*)(baseT + col * stride + kb + 16);
  return __builtin_shufflevector(lo, hi, 0,1,2,3,4,5,6,7,8,9,10,11,12,13,14,15);
}
__device__ inline v8f wmma_bf16(v16bf a, v16bf b, v8f c) {
  return __builtin_amdgcn_wmma_f32_16x16x32_bf16(false, a, false, b, (short)0, c, false, false);
}
// async 16B memory->LDS copy (ASYNCcnt-tracked, CDNA5 path)
__device__ inline void async_copy_b128(unsigned lds_off, const bf16_t* gsrc) {
  asm volatile("global_load_async_to_lds_b128 %0, %1, off"
               :: "v"(lds_off), "v"(gsrc) : "memory");
}
__device__ inline void wait_async0() {
  asm volatile("s_wait_asynccnt 0x0" ::: "memory");
}

#define LDS_STRIDE 40  // 32 + 8 pad (80B rows: 16B aligned, bank-spread)
#define BM 128
#define BN 64
#define BK 32

// ---------------------------------------------------------------- FFN1: gather tokens * w1 -> gelu -> h (bf16)
__global__ __launch_bounds__(256) void ffn1_kernel(
    const bf16_t* __restrict__ xb, const float* __restrict__ w1,
    const float* __restrict__ b1, const int* __restrict__ list,
    const int* __restrict__ counts, const int* __restrict__ offsets,
    bf16_t* __restrict__ hbuf) {
  const int e = blockIdx.z;
  const int cnt = counts[e];
  const int m0 = blockIdx.y * BM;
  if (m0 >= cnt) return;
  const int n0 = blockIdx.x * BN;
  const int off = offsets[e];
  const float* w1e = w1 + (size_t)e * H_DIM * I_DIM;

  __shared__ __align__(16) bf16_t Asm[BM * LDS_STRIDE];
  __shared__ __align__(16) bf16_t Bsm[BN * LDS_STRIDE];

  const int tid = threadIdx.x;
  const int lane = tid & 31, wid = tid >> 5;
  const int wm = wid >> 1, wn = wid & 1;   // wave tile: rows wm*32..+31, cols wn*32..+31

  v8f c00 = {}, c01 = {}, c10 = {}, c11 = {};

  // A staging: 128x32 bf16 tile = 512 x 16B chunks, 2 per thread; rows gathered
  const bf16_t* agsrc[2];
  unsigned alds[2];
  #pragma unroll
  for (int q = 0; q < 2; q++) {
    const int chunk = tid + q * 256;
    const int row = chunk >> 2;          // 0..127
    const int kq = (chunk & 3) * 8;      // 0,8,16,24
    int idx = off + m0 + row;            // clamp: list[] fully populated -> finite data
    if (idx > NSLOT - 1) idx = NSLOT - 1;
    const int tok = list[idx];
    agsrc[q] = xb + (size_t)tok * H_DIM + kq;
    alds[q] = (unsigned)(uintptr_t)&Asm[row * LDS_STRIDE + kq];
  }

  for (int k0 = 0; k0 < H_DIM; k0 += BK) {
    __syncthreads();
    #pragma unroll
    for (int q = 0; q < 2; q++) async_copy_b128(alds[q], agsrc[q] + k0);
    // B tile (fp32 -> bf16, transposed into LDS): 64x32 = 512 float4, 2 per thread
    #pragma unroll
    for (int rep = 0; rep < 2; rep++) {
      const int lin = tid + rep * 256;
      const int kk = lin >> 4;             // 0..31
      const int nn = (lin & 15) * 4;       // 0..60
      const float* src = &w1e[(size_t)(k0 + kk) * I_DIM + n0 + nn];
      const v4f v = *(const v4f*)src;
      Bsm[(nn + 0) * LDS_STRIDE + kk] = (bf16_t)v.x;
      Bsm[(nn + 1) * LDS_STRIDE + kk] = (bf16_t)v.y;
      Bsm[(nn + 2) * LDS_STRIDE + kk] = (bf16_t)v.z;
      Bsm[(nn + 3) * LDS_STRIDE + kk] = (bf16_t)v.w;
      __builtin_prefetch(src + (size_t)BK * I_DIM, 0, 1);
    }
    wait_async0();
    __syncthreads();

    const v16bf a0 = load_a_frag(Asm, LDS_STRIDE, wm * 32 + (lane & 15), lane);
    const v16bf a1 = load_a_frag(Asm, LDS_STRIDE, wm * 32 + 16 + (lane & 15), lane);
    const v16bf b0 = load_b_frag(Bsm, LDS_STRIDE, wn * 32 + (lane & 15), lane);
    const v16bf b1f = load_b_frag(Bsm, LDS_STRIDE, wn * 32 + 16 + (lane & 15), lane);
    c00 = wmma_bf16(a0, b0, c00);
    c01 = wmma_bf16(a0, b1f, c01);
    c10 = wmma_bf16(a1, b0, c10);
    c11 = wmma_bf16(a1, b1f, c11);
  }

  const float* b1e = b1 + (size_t)e * I_DIM;
  const int nc0 = n0 + wn * 32 + (lane & 15);
  const int nc1 = nc0 + 16;
  #pragma unroll
  for (int sm = 0; sm < 2; sm++) {
    const int mbase = m0 + wm * 32 + sm * 16 + (lane >> 4) * 8;
    #pragma unroll
    for (int i = 0; i < 8; i++) {
      const int m = mbase + i;
      if (m < cnt) {
        const size_t slot = (size_t)(off + m);
        float v0 = (sm ? c10[i] : c00[i]) + b1e[nc0];
        v0 = 0.5f * v0 * (1.0f + erff(v0 * 0.70710678118654752f));
        hbuf[slot * I_DIM + nc0] = (bf16_t)v0;
        float v1 = (sm ? c11[i] : c01[i]) + b1e[nc1];
        v1 = 0.5f * v1 * (1.0f + erff(v1 * 0.70710678118654752f));
        hbuf[slot * I_DIM + nc1] = (bf16_t)v1;
      }
    }
  }
}

// ---------------------------------------------------------------- FFN2: h * w2 + b2 + residual -> y (fp32)
__global__ __launch_bounds__(256) void ffn2_kernel(
    const bf16_t* __restrict__ hbuf, const float* __restrict__ w2,
    const float* __restrict__ b2, const float* __restrict__ x,
    const int* __restrict__ list, const int* __restrict__ counts,
    const int* __restrict__ offsets, float* __restrict__ ybuf) {
  const int e = blockIdx.z;
  const int cnt = counts[e];
  const int m0 = blockIdx.y * BM;
  if (m0 >= cnt) return;
  const int n0 = blockIdx.x * BN;
  const int off = offsets[e];
  const float* w2e = w2 + (size_t)e * I_DIM * H_DIM;

  __shared__ __align__(16) bf16_t Asm[BM * LDS_STRIDE];
  __shared__ __align__(16) bf16_t Bsm[BN * LDS_STRIDE];

  const int tid = threadIdx.x;
  const int lane = tid & 31, wid = tid >> 5;
  const int wm = wid >> 1, wn = wid & 1;

  v8f c00 = {}, c01 = {}, c10 = {}, c11 = {};

  const bf16_t* agsrc[2];
  unsigned alds[2];
  #pragma unroll
  for (int q = 0; q < 2; q++) {
    const int chunk = tid + q * 256;
    const int row = chunk >> 2;
    const int kq = (chunk & 3) * 8;
    int idx = off + m0 + row;            // clamp: hbuf rows all written by ffn1
    if (idx > NSLOT - 1) idx = NSLOT - 1;
    agsrc[q] = hbuf + (size_t)idx * I_DIM + kq;
    alds[q] = (unsigned)(uintptr_t)&Asm[row * LDS_STRIDE + kq];
  }

  for (int k0 = 0; k0 < I_DIM; k0 += BK) {
    __syncthreads();
    #pragma unroll
    for (int q = 0; q < 2; q++) async_copy_b128(alds[q], agsrc[q] + k0);
    #pragma unroll
    for (int rep = 0; rep < 2; rep++) {
      const int lin = tid + rep * 256;
      const int kk = lin >> 4;
      const int nn = (lin & 15) * 4;
      const float* src = &w2e[(size_t)(k0 + kk) * H_DIM + n0 + nn];
      const v4f v = *(const v4f*)src;
      Bsm[(nn + 0) * LDS_STRIDE + kk] = (bf16_t)v.x;
      Bsm[(nn + 1) * LDS_STRIDE + kk] = (bf16_t)v.y;
      Bsm[(nn + 2) * LDS_STRIDE + kk] = (bf16_t)v.z;
      Bsm[(nn + 3) * LDS_STRIDE + kk] = (bf16_t)v.w;
      __builtin_prefetch(src + (size_t)BK * H_DIM, 0, 1);
    }
    wait_async0();
    __syncthreads();

    const v16bf a0 = load_a_frag(Asm, LDS_STRIDE, wm * 32 + (lane & 15), lane);
    const v16bf a1 = load_a_frag(Asm, LDS_STRIDE, wm * 32 + 16 + (lane & 15), lane);
    const v16bf b0 = load_b_frag(Bsm, LDS_STRIDE, wn * 32 + (lane & 15), lane);
    const v16bf b1f = load_b_frag(Bsm, LDS_STRIDE, wn * 32 + 16 + (lane & 15), lane);
    c00 = wmma_bf16(a0, b0, c00);
    c01 = wmma_bf16(a0, b1f, c01);
    c10 = wmma_bf16(a1, b0, c10);
    c11 = wmma_bf16(a1, b1f, c11);
  }

  const float* b2e = b2 + (size_t)e * H_DIM;
  const int nc0 = n0 + wn * 32 + (lane & 15);
  const int nc1 = nc0 + 16;
  #pragma unroll
  for (int sm = 0; sm < 2; sm++) {
    const int mbase = m0 + wm * 32 + sm * 16 + (lane >> 4) * 8;
    #pragma unroll
    for (int i = 0; i < 8; i++) {
      const int m = mbase + i;
      if (m < cnt) {
        const size_t slot = (size_t)(off + m);
        const int t = list[slot];
        const float* xr = x + (size_t)t * H_DIM;
        ybuf[slot * H_DIM + nc0] = (sm ? c10[i] : c00[i]) + b2e[nc0] + xr[nc0];
        ybuf[slot * H_DIM + nc1] = (sm ? c11[i] : c01[i]) + b2e[nc1] + xr[nc1];
      }
    }
  }
}

// ---------------------------------------------------------------- per-token LayerNorm + weighted combine
__global__ __launch_bounds__(256) void combine_kernel(
    const float* __restrict__ y, const int* __restrict__ slot_of,
    const int* __restrict__ sel, const float* __restrict__ wnorm,
    const float* __restrict__ ln_g, const float* __restrict__ ln_b,
    float* __restrict__ out) {
  const int t = blockIdx.x;
  const int tid = threadIdx.x;
  const int lane = tid & 31, wid = tid >> 5;
  __shared__ float red[16];
  float acc[4] = {0.f, 0.f, 0.f, 0.f};

  for (int j = 0; j < K_TOP; j++) {
    const int slot = slot_of[t * 2 + j];
    const int e = sel[t * 2 + j];
    const float w = wnorm[t * 2 + j];
    const float* yr = y + (size_t)slot * H_DIM;
    float v[4];
    float s = 0.f, sq = 0.f;
    #pragma unroll
    for (int q = 0; q < 4; q++) {
      v[q] = yr[tid + q * 256];
      s += v[q];
      sq += v[q] * v[q];
    }
    for (int m = 16; m >= 1; m >>= 1) {
      s += __shfl_xor(s, m, 32);
      sq += __shfl_xor(sq, m, 32);
    }
    if (lane == 0) { red[wid] = s; red[8 + wid] = sq; }
    __syncthreads();
    if (wid == 0) {
      float a = (lane < 8) ? red[lane] : 0.f;
      float b = (lane < 8) ? red[8 + lane] : 0.f;
      for (int m = 4; m >= 1; m >>= 1) {
        a += __shfl_xor(a, m, 32);
        b += __shfl_xor(b, m, 32);
      }
      if (lane == 0) { red[0] = a; red[8] = b; }
    }
    __syncthreads();
    const float mu = red[0] * (1.f / (float)H_DIM);
    const float var = red[8] * (1.f / (float)H_DIM) - mu * mu;
    const float rstd = rsqrtf(var + LN_EPS);
    const float* g = ln_g + (size_t)e * H_DIM;
    const float* bb = ln_b + (size_t)e * H_DIM;
    #pragma unroll
    for (int q = 0; q < 4; q++) {
      const int hidx = tid + q * 256;
      acc[q] += w * ((v[q] - mu) * rstd * g[hidx] + bb[hidx]);
    }
    __syncthreads();
  }
  #pragma unroll
  for (int q = 0; q < 4; q++) out[(size_t)t * H_DIM + tid + q * 256] = acc[q];
}

// ---------------------------------------------------------------- launch
extern "C" void kernel_launch(void* const* d_in, const int* in_sizes, int n_in,
                              void* d_out, int out_size, void* d_ws, size_t ws_size,
                              hipStream_t stream) {
  (void)in_sizes; (void)n_in; (void)out_size; (void)ws_size;
  const float* x        = (const float*)d_in[0];
  const float* router_w = (const float*)d_in[1];
  const float* w1       = (const float*)d_in[2];
  const float* b1       = (const float*)d_in[3];
  const float* w2       = (const float*)d_in[4];
  const float* b2       = (const float*)d_in[5];
  const float* ln_g     = (const float*)d_in[6];
  const float* ln_b     = (const float*)d_in[7];
  float* out = (float*)d_out;
  float* logits_out = out + (size_t)T_TOK * H_DIM;

  char* p = (char*)d_ws;
  auto alloc = [&](size_t bytes) -> char* {
    char* r = p;
    p += (bytes + 255) & ~(size_t)255;
    return r;
  };
  int*    counts  = (int*)alloc(E_EXP * 4);
  int*    cursor  = (int*)alloc(E_EXP * 4);
  int*    offsets = (int*)alloc(E_EXP * 4);
  int*    sel     = (int*)alloc((size_t)NSLOT * 4);
  float*  wnorm   = (float*)alloc((size_t)NSLOT * 4);
  int*    slot_of = (int*)alloc((size_t)NSLOT * 4);
  int*    list    = (int*)alloc((size_t)NSLOT * 4);
  bf16_t* xb      = (bf16_t*)alloc((size_t)T_TOK * H_DIM * 2);
  bf16_t* hbuf    = (bf16_t*)alloc((size_t)NSLOT * I_DIM * 2);
  float*  ybuf    = (float*)alloc((size_t)NSLOT * H_DIM * 4);

  hipMemsetAsync(counts, 0, E_EXP * 4, stream);
  hipMemsetAsync(cursor, 0, E_EXP * 4, stream);

  cvt_bf16_kernel<<<(T_TOK * H_DIM) / (256 * 4), 256, 0, stream>>>(x, xb);
  router_kernel<<<T_TOK / 8, 256, 0, stream>>>(x, router_w, logits_out, sel, wnorm, counts);
  scan_kernel<<<1, 32, 0, stream>>>(counts, offsets);
  assign_kernel<<<T_TOK / 256, 256, 0, stream>>>(sel, offsets, cursor, list, slot_of);

  dim3 g1(I_DIM / BN, T_TOK / BM, E_EXP);
  ffn1_kernel<<<g1, 256, 0, stream>>>(xb, w1, b1, list, counts, offsets, hbuf);
  dim3 g2(H_DIM / BN, T_TOK / BM, E_EXP);
  ffn2_kernel<<<g2, 256, 0, stream>>>(hbuf, w2, b2, x, list, counts, offsets, ybuf);

  combine_kernel<<<T_TOK, 256, 0, stream>>>(ybuf, slot_of, sel, wnorm, ln_g, ln_b, out);
}
